// MlpKnnGnn_89644557403165
// MI455X (gfx1250) — matrix operand order repaired and checked by reading, not compiled, not run on hardware
//
#include <hip/hip_runtime.h>
#include <math.h>

// ---------------- CDNA5 WMMA types ----------------
typedef __attribute__((ext_vector_type(16))) __bf16       v16bf;
typedef __attribute__((ext_vector_type(8)))  float        v8f;
typedef __attribute__((ext_vector_type(4)))  float        f32x4;
typedef __attribute__((ext_vector_type(4)))  unsigned int u32x4;

#define D    256
#define KNN  32
#define SK   264          // padded LDS K-stride in bf16 elems (16B-aligned rows, bank-spread)
#define CB   64           // distance-kernel column block

union Frag {
    v16bf v;
    u32x4 u[2];
};

// Load one 16x32 (A) or 32x16-as-NT (B) bf16 fragment: two 16-byte chunks per lane.
__device__ __forceinline__ void load_frag(Frag& f, const __bf16* p0, const __bf16* p1) {
    f.u[0] = *(const u32x4*)p0;
    f.u[1] = *(const u32x4*)p1;
}

// 8 B-fragments (one 16-col tile, full K=256) from an N-major bf16 image with row stride `stride`.
__device__ __forceinline__ void load_btile(Frag* f, const __bf16* Bt, size_t stride,
                                           int nt, int ln, int hl) {
#pragma unroll
    for (int kk = 0; kk < 8; ++kk) {
        const __bf16* bb = Bt + (size_t)(nt * 16 + ln) * stride + kk * 32 + hl * 8;
        load_frag(f[kk], bb, bb + 16);
    }
}

// 8 chained WMMAs: full K=256 for one 16x16 tile.
__device__ __forceinline__ v8f wmma_tile(const Frag* a, const Frag* b) {
    v8f acc = {};
#pragma unroll
    for (int kk = 0; kk < 8; ++kk)
        acc = __builtin_amdgcn_wmma_f32_16x16x32_bf16(false, a[kk].v, false, b[kk].v,
                                                      (short)0, acc, false, false);
    return acc;
}

// ---------------- CDNA5 async global->LDS (ASYNCcnt) ----------------
__device__ __forceinline__ void async_ld16(unsigned int lds_addr, const void* gaddr) {
    asm volatile("global_load_async_to_lds_b128 %0, %1, off"
                 :: "v"(lds_addr), "v"((unsigned long long)(size_t)gaddr)
                 : "memory");
}
__device__ __forceinline__ void wait_async0() {
#if __has_builtin(__builtin_amdgcn_s_wait_asynccnt)
    __builtin_amdgcn_s_wait_asynccnt(0);
#else
    asm volatile("s_wait_asynccnt 0x0" ::: "memory");
#endif
}

// ---------------- conversion kernels ----------------
__global__ void k_cvt_bf16(const float* __restrict__ src, __bf16* __restrict__ dst, int n) {
    int i = blockIdx.x * blockDim.x + threadIdx.x;
    if (i < n) dst[i] = (__bf16)src[i];
}

// Wt[n*D + k] = W[k*D + n]  (N-major image of B so WMMA B-fragments are contiguous in K)
__global__ void k_cvt_wT(const float* __restrict__ W, __bf16* __restrict__ Wt) {
    int n = blockIdx.x;
    for (int k = threadIdx.x; k < D; k += blockDim.x)
        Wt[n * D + k] = (__bf16)W[k * D + n];
}

// ---------------- WMMA GEMM core: 16 col tiles, single B tile set (no spills), B from LDS ----
template <bool FULL>
__device__ __forceinline__ void gemm_tiles(const Frag* af, const __bf16* Blds,
                                           const float* __restrict__ bias,
                                           __bf16* __restrict__ Cout,
                                           int growbase, int ln, int hl, int n_rows, int relu) {
#pragma unroll 1
    for (int t = 0; t < 16; ++t) {
        Frag b[8];
        load_btile(b, Blds, SK, t, ln, hl);     // 16x ds_load_b128 clause
        v8f acc = wmma_tile(af, b);             // 8 back-to-back WMMAs
        int col = t * 16 + ln;
        float bv = bias[col];
#pragma unroll
        for (int r = 0; r < 8; ++r) {
            int row = growbase + r + hl * 8;
            if (FULL || row < n_rows) {
                float v = acc[r] + bv;
                if (relu) v = fmaxf(v, 0.0f);
                Cout[(size_t)row * D + col] = (__bf16)v;
            }
        }
    }
}

// C = act(A[bf16, NxD] @ B + bias), B supplied as Bt[n][k] (N-major). A and Bt staged in LDS.
__global__ void __launch_bounds__(256, 1)
k_gemm_bias(const __bf16* __restrict__ A, const __bf16* __restrict__ Bt,
            const float* __restrict__ bias, __bf16* __restrict__ Cout,
            int n_rows, int relu) {
    extern __shared__ char smem[];
    __bf16* Alds = (__bf16*)smem;                 // 128 x SK bf16
    __bf16* Blds = Alds + 128 * SK;               // 256 x SK bf16

    const int tid  = threadIdx.x;
    const int wv   = tid >> 5;
    const int lane = tid & 31;
    const int hl   = lane >> 4;
    const int ln   = lane & 15;
    const int mblk = blockIdx.x * 128;

    // cooperative load: 128 rows x 256 K of A into LDS (zero-fill past n_rows)
    {
        int row  = tid >> 1;
        int kc   = (tid & 1) * 128;
        u32x4* l = (u32x4*)(Alds + row * SK + kc);
        int grow = mblk + row;
        if (grow < n_rows) {
            const u32x4* g = (const u32x4*)(A + (size_t)grow * D + kc);
#pragma unroll
            for (int i = 0; i < 16; ++i) l[i] = g[i];
        } else {
            u32x4 z = (u32x4)0u;
#pragma unroll
            for (int i = 0; i < 16; ++i) l[i] = z;
        }
    }
    // cooperative load: full Bt (256 x 256) into LDS, one row per thread
    {
        const u32x4* g = (const u32x4*)(Bt + (size_t)tid * D);
        u32x4* l = (u32x4*)(Blds + tid * SK);
#pragma unroll
        for (int i = 0; i < 32; ++i) l[i] = g[i];
    }
    __syncthreads();

    const int mrow = wv * 16;
    Frag af[8];
#pragma unroll
    for (int kk = 0; kk < 8; ++kk) {
        const __bf16* base = Alds + (mrow + ln) * SK + kk * 32;
        load_frag(af[kk], base + hl * 8, base + 16 + hl * 8);
    }

    const int growbase = mblk + mrow;
    if (mblk + 128 <= n_rows)
        gemm_tiles<true >(af, Blds, bias, Cout, growbase, ln, hl, n_rows, relu);
    else
        gemm_tiles<false>(af, Blds, bias, Cout, growbase, ln, hl, n_rows, relu);
}

// ---------------- row squared norms ----------------
__global__ void k_rowsq(const __bf16* __restrict__ H, float* __restrict__ sq, int n) {
    int i = blockIdx.x * blockDim.x + threadIdx.x;
    if (i >= n) return;
    float s = 0.f;
    for (int d = 0; d < D; ++d) { float v = (float)H[(size_t)i * D + d]; s += v * v; }
    sq[i] = s;
}

// ---------------- fused WMMA distance + top-k(32) + edge emit ----------------
// d_out layout (floats): [0,N) out | [N, N+Nk) src(int) | [N+Nk, N+2Nk) dst(int) | [N+2Nk, N+3Nk) attr
// colb is double-buffered in LDS; next block staged with global_load_async_to_lds_b128 (ASYNCcnt).
__device__ __forceinline__ void issue_colblock(const __bf16* __restrict__ H, __bf16* bufp,
                                               int cb, int n_rows, int tid) {
    int col  = tid >> 2;
    int kc   = (tid & 3) * 64;
    int gcol = cb * CB + col;
    if (gcol >= n_rows) gcol = n_rows - 1;         // clamp: OOB cols excluded via sq sentinel
    const __bf16* g = H + (size_t)gcol * D + kc;
    unsigned int l  = (unsigned int)(size_t)(bufp + col * SK + kc);
#pragma unroll
    for (int i = 0; i < 8; ++i)
        async_ld16(l + i * 16, g + i * 8);
}

__global__ void __launch_bounds__(256, 1)
k_dist_topk(const __bf16* __restrict__ H, const float* __restrict__ sq,
            float* __restrict__ dout, float* __restrict__ deg, int n_rows) {
    extern __shared__ char smem[];
    __bf16* colb0 = (__bf16*)smem;                               // CB x SK bf16
    __bf16* colb1 = colb0 + CB * SK;                             // CB x SK bf16
    float*  stage = (float*)(smem + (size_t)2 * CB * SK * 2);    // 8 waves * 16 * CB
    float*  tkd   = stage + 8 * 16 * CB;                         // 128 * KNN
    int*    tki   = (int*)(tkd + 128 * KNN);                     // 128 * KNN

    const int tid  = threadIdx.x;
    const int wv   = tid >> 5;
    const int lane = tid & 31;
    const int hl   = lane >> 4;
    const int ln   = lane & 15;
    const int mblk = blockIdx.x * 128;
    const int wbase = mblk + wv * 16;
    const bool waveActive = (wbase < n_rows);

    for (int t = tid; t < 128 * KNN; t += blockDim.x) { tkd[t] = 3.0e38f; tki[t] = 0; }

    // register-resident A fragments: this wave's 16 rows, full K=256
    Frag af[8];
    int arow = waveActive ? (wbase + ln) : 0;
#pragma unroll
    for (int kk = 0; kk < 8; ++kk) {
        const __bf16* base = H + (size_t)arow * D + kk * 32;
        load_frag(af[kk], base + hl * 8, base + 16 + hl * 8);
    }
    float sqa[8];
#pragma unroll
    for (int r = 0; r < 8; ++r) {
        int rr = wbase + r + hl * 8;
        sqa[r] = (rr < n_rows) ? sq[rr] : 0.f;
    }

    float* stw   = stage + wv * 16 * CB;
    float* mytkd = tkd + (wv * 16 + ln) * KNN;
    int*   mytki = tki + (wv * 16 + ln) * KNN;

    const int nblocks = (n_rows + CB - 1) / CB;
    issue_colblock(H, colb0, 0, n_rows, tid);                    // prologue async stage

    for (int cb = 0; cb < nblocks; ++cb) {
        __bf16* cbuf = (cb & 1) ? colb1 : colb0;
        __bf16* nbuf = (cb & 1) ? colb0 : colb1;

        wait_async0();            // this wave's staged block has landed in LDS
        __syncthreads();          // all waves' stages landed; prior consumers of nbuf done
        if (cb + 1 < nblocks) issue_colblock(H, nbuf, cb + 1, n_rows, tid);
        if (!waveActive) continue;

#pragma unroll 1
        for (int nt = 0; nt < CB / 16; ++nt) {
            Frag bf8[8];
            load_btile(bf8, cbuf, SK, nt, ln, hl);
            v8f acc = wmma_tile(af, bf8);
            int gj = cb * CB + nt * 16 + ln;
            float sb = (gj < n_rows) ? sq[gj] : 3.0e38f;         // sentinel excludes OOB cols
#pragma unroll
            for (int r = 0; r < 8; ++r) {
                float d2 = sqa[r] + sb - 2.0f * acc[r];
                stw[(r + hl * 8) * CB + nt * 16 + ln] = sqrtf(fmaxf(d2, 0.0f));
            }
        }

        // top-k update: lanes 0-15 each own one row; b128 candidate reads + cached threshold
        if (hl == 0) {
            int grow = wbase + ln;
            if (grow < n_rows) {
                float thr = mytkd[KNN - 1];
                const float* srow = stw + ln * CB;
                for (int c = 0; c < CB; c += 4) {
                    f32x4 dv = *(const f32x4*)(srow + c);
#pragma unroll
                    for (int q = 0; q < 4; ++q) {
                        float dc = dv[q];
                        if (dc < thr) {
                            int j = cb * CB + c + q;
                            int p = KNN - 1;
                            while (p > 0 && mytkd[p - 1] > dc) {
                                mytkd[p] = mytkd[p - 1];
                                mytki[p] = mytki[p - 1];
                                --p;
                            }
                            mytkd[p] = dc;
                            mytki[p] = j;
                            thr = mytkd[KNN - 1];
                        }
                    }
                }
            }
        }
    }
    __syncthreads();

    // finalize: emit edges sorted by ascending distance (== top_k of -dist), accumulate degree
    if (waveActive && hl == 0) {
        int grow = wbase + ln;
        if (grow < n_rows) {
            size_t Nk   = (size_t)n_rows * KNN;
            int*   esrc = (int*)(dout + n_rows);
            int*   edst = (int*)(dout + n_rows) + Nk;
            float* eatt = dout + n_rows + 2 * Nk;
            float dsum = 0.f;
            for (int t = 0; t < KNN; ++t) {
                float w = expf(-mytkd[t]);
                size_t e = (size_t)grow * KNN + t;
                esrc[e] = mytki[t];
                edst[e] = grow;
                eatt[e] = w;
                dsum += w;
            }
            deg[grow] = dsum;
        }
    }
}

// ---------------- GCN layer 1 gather: agg = scatter(norm * feats[src]) -> bf16 ----------------
__global__ void k_gather1(const float* __restrict__ feats, const float* __restrict__ dro,
                          const float* __restrict__ deg, __bf16* __restrict__ aggb, int n_rows) {
    int node = blockIdx.x;
    int d    = threadIdx.x;                 // 256 dims
    size_t Nk = (size_t)n_rows * KNN;
    const int*   esrc = (const int*)(dro + n_rows);
    const float* eatt = dro + n_rows + 2 * Nk;
    float dinv_i = rsqrtf(fmaxf(deg[node], 1e-12f));
    float acc = 0.f;
    for (int t = 0; t < KNN; ++t) {
        size_t e = (size_t)node * KNN + t;
        int s = esrc[e];
        float nrm = eatt[e] * dinv_i * rsqrtf(fmaxf(deg[s], 1e-12f));
        acc += nrm * feats[(size_t)s * D + d];
    }
    aggb[(size_t)node * D + d] = (__bf16)acc;
}

// ---------------- y = x1 @ Wg2 (N x 256 x 1) ----------------
__global__ void k_gemv(const __bf16* __restrict__ X, const float* __restrict__ Wg2,
                       float* __restrict__ y, int n_rows) {
    int i = blockIdx.x * blockDim.x + threadIdx.x;
    if (i >= n_rows) return;
    float s = 0.f;
    for (int d = 0; d < D; ++d) s += (float)X[(size_t)i * D + d] * Wg2[d];
    y[i] = s;
}

// ---------------- out[i] = bg2 + sum_e norm_e * y[src_e]  (layer 2 collapsed to scalar gather) ----
__global__ void k_out(const float* __restrict__ y, const float* __restrict__ deg,
                      const float* __restrict__ dro, const float* __restrict__ bg2,
                      float* __restrict__ out, int n_rows) {
    int i = blockIdx.x * blockDim.x + threadIdx.x;
    if (i >= n_rows) return;
    size_t Nk = (size_t)n_rows * KNN;
    const int*   esrc = (const int*)(dro + n_rows);
    const float* eatt = dro + n_rows + 2 * Nk;
    float dinv_i = rsqrtf(fmaxf(deg[i], 1e-12f));
    float s = bg2[0];
    for (int t = 0; t < KNN; ++t) {
        size_t e = (size_t)i * KNN + t;
        int si = esrc[e];
        s += eatt[e] * dinv_i * rsqrtf(fmaxf(deg[si], 1e-12f)) * y[si];
    }
    out[i] = s;
}

// ---------------- host ----------------
extern "C" void kernel_launch(void* const* d_in, const int* in_sizes, int n_in,
                              void* d_out, int out_size, void* d_ws, size_t ws_size,
                              hipStream_t stream) {
    const float* feats = (const float*)d_in[0];
    const float* W1    = (const float*)d_in[1];
    const float* b1    = (const float*)d_in[2];
    const float* W2    = (const float*)d_in[3];
    const float* b2    = (const float*)d_in[4];
    const float* Wg1   = (const float*)d_in[5];
    const float* bg1   = (const float*)d_in[6];
    const float* Wg2   = (const float*)d_in[7];
    const float* bg2   = (const float*)d_in[8];
    const int N = in_sizes[0] / D;

    char* ws = (char*)d_ws;
    size_t off = 0;
    auto alloc = [&](size_t bytes) -> char* {
        char* p = ws + off;
        off += (bytes + 255) & ~(size_t)255;
        return p;
    };
    __bf16* featsb = (__bf16*)alloc((size_t)N * D * 2);
    __bf16* h1b    = (__bf16*)alloc((size_t)N * D * 2);
    __bf16* hb     = (__bf16*)alloc((size_t)N * D * 2);
    __bf16* aggb   = (__bf16*)alloc((size_t)N * D * 2);
    __bf16* x1b    = (__bf16*)alloc((size_t)N * D * 2);
    __bf16* W1t    = (__bf16*)alloc((size_t)D * D * 2);
    __bf16* W2t    = (__bf16*)alloc((size_t)D * D * 2);
    __bf16* Wg1t   = (__bf16*)alloc((size_t)D * D * 2);
    float*  sq     = (float*)alloc((size_t)N * 4);
    float*  deg    = (float*)alloc((size_t)N * 4);
    float*  yv     = (float*)alloc((size_t)N * 4);

    int total = N * D;
    k_cvt_bf16<<<(total + 255) / 256, 256, 0, stream>>>(feats, featsb, total);
    k_cvt_wT<<<D, 256, 0, stream>>>(W1, W1t);
    k_cvt_wT<<<D, 256, 0, stream>>>(W2, W2t);
    k_cvt_wT<<<D, 256, 0, stream>>>(Wg1, Wg1t);

    const size_t gemmLds = (size_t)(128 + 256) * SK * 2;   // A block + full Bt in LDS
    const int mblocks = (N + 127) / 128;
    k_gemm_bias<<<mblocks, 256, gemmLds, stream>>>(featsb, W1t, b1, h1b, N, 1);
    k_gemm_bias<<<mblocks, 256, gemmLds, stream>>>(h1b, W2t, b2, hb, N, 0);
    k_rowsq<<<(N + 255) / 256, 256, 0, stream>>>(hb, sq, N);

    const size_t distLds = (size_t)2 * CB * SK * 2 + 8 * 16 * CB * 4 + 128 * KNN * 8;
    k_dist_topk<<<mblocks, 256, distLds, stream>>>(hb, sq, (float*)d_out, deg, N);

    k_gather1<<<N, D, 0, stream>>>(feats, (const float*)d_out, deg, aggb, N);
    k_gemm_bias<<<mblocks, 256, gemmLds, stream>>>(aggb, Wg1t, bg1, x1b, N, 1);
    k_gemv<<<(N + 255) / 256, 256, 0, stream>>>(x1b, Wg2, yv, N);
    k_out<<<(N + 255) / 256, 256, 0, stream>>>(yv, deg, (const float*)d_out, bg2, (float*)d_out, N);
}
